// ResidualVQ_43293270344191
// MI455X (gfx1250) — matrix operand order repaired and checked by reading, not compiled, not run on hardware
//
#include <hip/hip_runtime.h>

typedef __attribute__((ext_vector_type(16))) __bf16 v16bf;
typedef __attribute__((ext_vector_type(8)))  float  v8f;
typedef __attribute__((ext_vector_type(4)))  float  v4f;

#define QL 8            // quantizer levels
#define BB 8            // batch
#define NN 4096         // rows per batch
#define DD 256          // feature dim
#define CC 1024         // codes per level
#define ROWS (BB*NN)    // 32768
#define WG_THREADS 256
#define WAVES 8
#define ROWS_PER_WAVE 16
#define ROWS_PER_WG (WAVES*ROWS_PER_WAVE)   // 128
#define NUM_WG (ROWS/ROWS_PER_WG)           // 256
#define WGS_PER_B (NN/ROWS_PER_WG)          // 32
#define RS 260          // LDS row stride (floats), 16B aligned, spreads banks
#define CT 64           // 16-code tiles per level
#define KB 8            // K blocks of 32 per d=256

// workspace layout (bytes)
#define WS_BHI  ((size_t)0)
#define WS_BLO  ((size_t)QL*CT*KB*1024)            // 4 MiB
#define WS_E2   ((size_t)2*QL*CT*KB*1024)          // 8 MiB
#define WS_LOSS (WS_E2 + (size_t)QL*CC*4)          // + 32 KiB
// loss partials: NUM_WG * WAVES * QL floats = 64 KiB

// WGP-scope prefetch (scope 0: pull into all cache levels, stop on WGP hit).
// __builtin_prefetch can only reach SE/DEV scope, which skips the WGP cache.
__device__ __forceinline__ void prefetch_wgp(const void* p) {
    asm volatile("global_prefetch_b8 %0, off" :: "v"(p));
}

// ---------------------------------------------------------------------------
// Prep 1: convert codebooks f32 -> bf16 hi/lo in WMMA B-operand tile layout.
// B (32x16, KxN) layout chosen here (A uses same k order): lanes 0-15 hold K=0..15 of column N=lane,
// lanes 16-31 hold K=16..31 of column N=lane-16; 16 bf16 (32B) per lane.
// Tile region per (level, ct, kb): 32 lanes * 32B = 1 KiB.
// ---------------------------------------------------------------------------
__global__ void rvq_prep_b(const float* __restrict__ cb, char* __restrict__ ws) {
    int tid  = blockIdx.x * blockDim.x + threadIdx.x;   // 131072 total
    int lane = tid & 31;
    int kb   = (tid >> 5) & 7;
    int ct   = (tid >> 8) & 63;
    int qi   = tid >> 14;
    int code = ct * 16 + (lane & 15);
    int k0   = kb * 32 + ((lane >> 4) << 4);
    const float* src = cb + ((size_t)(qi * CC + code) * DD + k0);
    v16bf hi, lo;
#pragma unroll
    for (int i = 0; i < 16; ++i) {
        float f   = src[i];
        __bf16 h  = (__bf16)f;
        hi[i]     = h;
        lo[i]     = (__bf16)(f - (float)h);
    }
    size_t off = ((size_t)(((qi * CT + ct) * KB + kb) * 32 + lane)) * 32;
    *(v16bf*)(ws + WS_BHI + off) = hi;
    *(v16bf*)(ws + WS_BLO + off) = lo;
}

// Prep 2: e2[q][c] = ||e||^2
__global__ void rvq_prep_e2(const float* __restrict__ cb, char* __restrict__ ws) {
    int tid = blockIdx.x * blockDim.x + threadIdx.x;    // 8192 total
    const float* src = cb + (size_t)tid * DD;
    float s = 0.f;
#pragma unroll 4
    for (int k = 0; k < DD; k += 4) {
        v4f v = *(const v4f*)(src + k);
        s += v[0]*v[0] + v[1]*v[1] + v[2]*v[2] + v[3]*v[3];
    }
    ((float*)(ws + WS_E2))[tid] = s;
}

// ---------------------------------------------------------------------------
// Main: per WG 128 rows (LDS-resident f32 residual), 8 levels in-kernel.
// Per wave: 16-row A tile in registers (bf16 hi/lo), stream B tiles from ws,
// 3x v_wmma_f32_16x16x32_bf16 per K-block, argmin via shuffles, LDS update.
// ---------------------------------------------------------------------------
__global__ __launch_bounds__(WG_THREADS)
void rvq_main(const float* __restrict__ x, const float* __restrict__ cb,
              const char* __restrict__ ws, float* __restrict__ qout,
              int* __restrict__ idx_out, float* __restrict__ lossP) {
    extern __shared__ float smem[];
    float* resid  = smem;                                   // 8*16*RS floats
    int*   idxbuf = (int*)(smem + WAVES * ROWS_PER_WAVE * RS);

    const int tid    = threadIdx.x;
    const int wave   = tid >> 5;
    const int lane   = tid & 31;
    const int rowl   = lane & 15;       // N column / A row within tile
    const int khalf  = lane >> 4;       // which K half this lane holds for A
    const int wgRow0 = blockIdx.x * ROWS_PER_WG;

    // stage x -> LDS residual
    for (int i = tid; i < ROWS_PER_WG * (DD / 4); i += WG_THREADS) {
        int r = i / (DD / 4), k4 = i % (DD / 4);
        v4f v = *(const v4f*)(x + (size_t)(wgRow0 + r) * DD + k4 * 4);
        *(v4f*)(resid + r * RS + k4 * 4) = v;
    }
    __syncthreads();

    const char*  BhiBase = ws + WS_BHI;
    const char*  BloBase = ws + WS_BLO;
    const float* e2Base  = (const float*)(ws + WS_E2);

    float waveLoss[QL];

    for (int lvl = 0; lvl < QL; ++lvl) {
        // ---- build A_hi / A_lo registers from LDS residual ----
        // A lane layout: lane l -> row (l&15); VGPR elems 0..15 = K kb*32+khalf*16..+15
        // (contiguous, matching the B tile layout).
        v16bf Ahi[KB], Alo[KB];
        const float* rrow = resid + (wave * 16 + rowl) * RS + khalf * 16;
#pragma unroll
        for (int kb = 0; kb < KB; ++kb) {
#pragma unroll
            for (int h = 0; h < 2; ++h) {
                v4f f0 = *(const v4f*)(rrow + kb * 32 + h * 8);
                v4f f1 = *(const v4f*)(rrow + kb * 32 + h * 8 + 4);
#pragma unroll
                for (int j = 0; j < 4; ++j) {
                    float f; __bf16 hb;
                    f = f0[j]; hb = (__bf16)f;
                    Ahi[kb][h * 8 + j] = hb;
                    Alo[kb][h * 8 + j] = (__bf16)(f - (float)hb);
                    f = f1[j]; hb = (__bf16)f;
                    Ahi[kb][h * 8 + 4 + j] = hb;
                    Alo[kb][h * 8 + 4 + j] = (__bf16)(f - (float)hb);
                }
            }
        }

        float bestv[8]; int bestc[8];
#pragma unroll
        for (int j = 0; j < 8; ++j) { bestv[j] = 3.4e38f; bestc[j] = 0; }

        const char*  bhL = BhiBase + (size_t)lvl * CT * KB * 1024;
        const char*  blL = BloBase + (size_t)lvl * CT * KB * 1024;
        const float* e2L = e2Base + lvl * CC;

        for (int ct = 0; ct < CT; ++ct) {
            const char* bh = bhL + (size_t)ct * KB * 1024 + lane * 32;
            const char* bl = blL + (size_t)ct * KB * 1024 + lane * 32;
            if (ct + 1 < CT) {
                // WGP-scope prefetch of the next code tile while this tile's
                // 24 WMMAs run (data already L2-resident; only WGP scope
                // actually fills the near cache).
                prefetch_wgp(bh + KB * 1024);
                prefetch_wgp(bl + KB * 1024);
            }
            v8f acc = {};
#pragma unroll
            for (int kb = 0; kb < KB; ++kb) {
                v16bf bhv = *(const v16bf*)(bh + kb * 1024);
                v16bf blv = *(const v16bf*)(bl + kb * 1024);
                acc = __builtin_amdgcn_wmma_f32_16x16x32_bf16(
                          false, Ahi[kb], false, bhv, (short)0, acc, false, false);
                acc = __builtin_amdgcn_wmma_f32_16x16x32_bf16(
                          false, Alo[kb], false, bhv, (short)0, acc, false, false);
                acc = __builtin_amdgcn_wmma_f32_16x16x32_bf16(
                          false, Ahi[kb], false, blv, (short)0, acc, false, false);
            }
            // score = ||e||^2 - 2 r.e  (||r||^2 constant over codes: dropped)
            float e2v = e2L[ct * 16 + rowl];
            int   code = ct * 16 + rowl;
#pragma unroll
            for (int j = 0; j < 8; ++j) {
                float s = e2v - 2.0f * acc[j];
                if (s < bestv[j]) { bestv[j] = s; bestc[j] = code; }
            }
        }

        // argmin across the 16 lanes holding each row (C layout: VGPR j,
        // lanes 0-15 -> row j; lanes 16-31 -> row j+8). Tie-break: lowest idx.
#pragma unroll
        for (int j = 0; j < 8; ++j) {
#pragma unroll
            for (int m = 1; m < 16; m <<= 1) {
                float ov = __shfl_xor(bestv[j], m, 16);
                int   oc = __shfl_xor(bestc[j], m, 16);
                if (ov < bestv[j] || (ov == bestv[j] && oc < bestc[j])) {
                    bestv[j] = ov; bestc[j] = oc;
                }
            }
        }
        if (rowl == 0) {     // lane 0 -> rows 0..7, lane 16 -> rows 8..15
#pragma unroll
            for (int j = 0; j < 8; ++j) {
                int rloc = wave * 16 + khalf * 8 + j;
                idxbuf[rloc] = bestc[j];
                idx_out[(size_t)(wgRow0 + rloc) * QL + lvl] = bestc[j];
            }
        }
        __syncthreads();

        // residual -= chosen code; loss = sum(new_residual^2)
        float lsum = 0.f;
        const float* cbL = cb + (size_t)lvl * CC * DD;
#pragma unroll 1
        for (int r = 0; r < ROWS_PER_WAVE; ++r) {
            int code = idxbuf[wave * 16 + r];
            const float* ev = cbL + (size_t)code * DD;
            float* rr = resid + (wave * 16 + r) * RS;
#pragma unroll
            for (int t = 0; t < 8; ++t) {
                int k = t * 32 + lane;
                float nv = rr[k] - ev[k];
                rr[k] = nv;
                lsum += nv * nv;
            }
        }
#pragma unroll
        for (int m = 1; m < 32; m <<= 1) lsum += __shfl_xor(lsum, m, 32);
        waveLoss[lvl] = lsum;
        __syncthreads();
    }

    // quantized_out = x - residual_final   (qout + residual == x invariant)
    for (int i = tid; i < ROWS_PER_WG * (DD / 4); i += WG_THREADS) {
        int r = i / (DD / 4), k4 = i % (DD / 4);
        v4f xv = *(const v4f*)(x + (size_t)(wgRow0 + r) * DD + k4 * 4);
        v4f rv = *(const v4f*)(resid + r * RS + k4 * 4);
        *(v4f*)(qout + (size_t)(wgRow0 + r) * DD + k4 * 4) = xv - rv;
    }
    if (lane == 0) {
        float* lp = lossP + ((size_t)blockIdx.x * WAVES + wave) * QL;
#pragma unroll
        for (int l = 0; l < QL; ++l) lp[l] = waveLoss[l];
    }
}

// Deterministic fixed-order loss reduction: losses[b,q] = sum(partials)/(n*d)
__global__ void rvq_loss_reduce(const float* __restrict__ lossP,
                                float* __restrict__ losses) {
    int t = threadIdx.x;
    if (t >= BB * QL) return;
    int b = t >> 3, q = t & 7;
    float s = 0.f;
    for (int wg = 0; wg < WGS_PER_B; ++wg)
        for (int wv = 0; wv < WAVES; ++wv)
            s += lossP[((size_t)(b * WGS_PER_B + wg) * WAVES + wv) * QL + q];
    losses[t] = s * (1.0f / ((float)NN * (float)DD));
}

extern "C" void kernel_launch(void* const* d_in, const int* in_sizes, int n_in,
                              void* d_out, int out_size, void* d_ws, size_t ws_size,
                              hipStream_t stream) {
    const float* x  = (const float*)d_in[0];   // [8,4096,256]
    const float* cb = (const float*)d_in[1];   // [8,1024,256]

    float* qout    = (float*)d_out;                                    // 8M f32
    int*   idx_out = (int*)((float*)d_out + (size_t)BB * NN * DD);     // 262144 i32
    float* losses  = (float*)d_out + (size_t)BB * NN * DD + (size_t)BB * NN * QL; // 64 f32

    char*  ws    = (char*)d_ws;
    float* lossP = (float*)(ws + WS_LOSS);

    rvq_prep_b<<<512, 256, 0, stream>>>(cb, ws);
    rvq_prep_e2<<<QL * CC / 256, 256, 0, stream>>>(cb, ws);

    size_t smem = (size_t)(WAVES * ROWS_PER_WAVE * RS) * sizeof(float)
                + (size_t)ROWS_PER_WG * sizeof(int);
    rvq_main<<<NUM_WG, WG_THREADS, smem, stream>>>(x, cb, ws, qout, idx_out, lossP);

    rvq_loss_reduce<<<1, 64, 0, stream>>>(lossP, losses);
}
